// DeepSeekSparseAttention_86406152061645
// MI455X (gfx1250) — matrix operand order repaired and checked by reading, not compiled, not run on hardware
//
#include <hip/hip_runtime.h>

// ---------------------------------------------------------------------------
// DeepSeek sparse attention for MI455X (gfx1250, wave32).
// Dense GEMMs: v_wmma_f32_16x16x32_f16 with double-buffered LDS tiles filled
// by the CDNA5 async copy engine (global_load_async_to_lds_b128 / ASYNCcnt,
// via inline asm since no builtin covers it). Operands are pre-converted to
// f16 (weights pre-transposed to [N][K]) by trivial one-shot kernels so the
// GEMM hot loop contains no format conversion: 3 async-copy issues for the
// next k-tile, s_wait_asynccnt, barrier, 8x ds_load_b128, 4x WMMA, barrier.
// Irregular stages (top-k select, gathered per-query attention) remain
// VALU/LDS code: per-query gather indices make them GEMV-shaped and they are
// ~10% of total FLOPs.
// ---------------------------------------------------------------------------

typedef __attribute__((ext_vector_type(16))) _Float16 v16h;
typedef __attribute__((ext_vector_type(8)))  _Float16 v8h;
typedef __attribute__((ext_vector_type(8)))  float    v8f;

#define S_LEN   2048
#define D_MODEL 512
#define NH      8
#define HD      64
#define RANK    64
#define TOPK    128

// LDS tile geometry: rows padded to 40 f16 (80 B stride; multiple of 16 B,
// 20-bank stride => conflict-free 16-B fragment reads).
#define TA_ELEMS (128 * 40)
#define TB_ELEMS (64 * 40)
#define TA_BYTES (TA_ELEMS * 2)

static __device__ __forceinline__ v16h frag_join(v8h lo, v8h hi) {
  v16h r;
#pragma unroll
  for (int i = 0; i < 8; ++i) { r[i] = lo[i]; r[i + 8] = hi[i]; }
  return r;
}

// CDNA5 async copy: per-lane 16 B, global -> LDS, tracked by ASYNCcnt.
static __device__ __forceinline__ void async_b128(unsigned lds_off,
                                                 const _Float16* g) {
  asm volatile("global_load_async_to_lds_b128 %0, %1, off"
               :: "v"(lds_off), "v"(g) : "memory");
}
static __device__ __forceinline__ void wait_async3() {
  asm volatile("s_wait_asynccnt 0x3" ::: "memory");
}
static __device__ __forceinline__ void wait_async0() {
  asm volatile("s_wait_asynccnt 0x0" ::: "memory");
}

// ---------------------------------------------------------------------------
// C[M,N] = alpha * A[M,K] * Bt[N,K]^T.  A, Bt are f16 row-major.
// Block tile 128(M) x 64(N), k-step 32, 256 threads = 8 waves (4x2 grid of
// 32x32 wave patches, 2x2 WMMA accumulators each). Double-buffered LDS.
// ---------------------------------------------------------------------------
__global__ void __launch_bounds__(256) wmma_gemm_f16(
    const _Float16* __restrict__ A, const _Float16* __restrict__ Bt,
    float* __restrict__ C, int M, int N, int K,
    int lda, int ldb, int ldc, float alpha)
{
  __shared__ __align__(16) _Float16 smem[2][TA_ELEMS + TB_ELEMS];

  const int tid    = threadIdx.x;
  const int blockM = blockIdx.y * 128;
  const int blockN = blockIdx.x * 64;

  const int wave = tid >> 5;
  const int lane = tid & 31;
  const int wm   = (wave >> 1) * 32;   // 0,32,64,96
  const int wn   = (wave & 1) * 32;    // 0,32
  const int lr   = lane & 15;
  const int lh   = lane >> 4;

  const unsigned sbase[2] = { (unsigned)(size_t)&smem[0][0],
                              (unsigned)(size_t)&smem[1][0] };

  v8f acc[2][2];
#pragma unroll
  for (int i = 0; i < 2; ++i)
#pragma unroll
    for (int j = 0; j < 2; ++j)
      acc[i][j] = (v8f){0.f, 0.f, 0.f, 0.f, 0.f, 0.f, 0.f, 0.f};

  // Per k-tile async stage: A = 128 rows x 64 B (512 lane-tasks -> 2 issues
  // per thread), B = 64 rows x 64 B (256 lane-tasks -> 1 issue per thread).
  const int ar0 = tid >> 2;          // 0..63
  const int seg = tid & 3;           // 16-B segment within a 64-B row
  auto issue_tile = [&](int k0, int buf) {
    const unsigned sb = sbase[buf];
    async_b128(sb + (unsigned)(ar0 * 80 + seg * 16),
               A + (size_t)(blockM + ar0) * lda + k0 + seg * 8);
    async_b128(sb + (unsigned)((ar0 + 64) * 80 + seg * 16),
               A + (size_t)(blockM + ar0 + 64) * lda + k0 + seg * 8);
    async_b128(sb + TA_BYTES + (unsigned)(ar0 * 80 + seg * 16),
               Bt + (size_t)(blockN + ar0) * ldb + k0 + seg * 8);
  };

  issue_tile(0, 0);
  const int nk = K >> 5;
  for (int kt = 0; kt < nk; ++kt) {
    if (kt + 1 < nk) {
      issue_tile((kt + 1) * 32, (kt + 1) & 1);  // prefetch next tile
      wait_async3();                            // current tile's 3 copies done
    } else {
      wait_async0();
    }
    __syncthreads();                            // all waves' copies visible

    const _Float16 (*As)[40] = (const _Float16 (*)[40])&smem[kt & 1][0];
    const _Float16 (*Bs)[40] = (const _Float16 (*)[40])&smem[kt & 1][TA_ELEMS];

    v16h af[2], bfr[2];
#pragma unroll
    for (int i = 0; i < 2; ++i) {
      const int row = wm + i * 16 + lr;
      const v8h lo = *(const v8h*)&As[row][8 * lh];        // K = 8h..8h+7
      const v8h hi = *(const v8h*)&As[row][16 + 8 * lh];   // K = 16+8h..
      af[i] = frag_join(lo, hi);
    }
#pragma unroll
    for (int j = 0; j < 2; ++j) {
      const int col = wn + j * 16 + lr;
      const v8h lo = *(const v8h*)&Bs[col][8 * lh];        // K = 8h..8h+7
      const v8h hi = *(const v8h*)&Bs[col][16 + 8 * lh];   // K = 16+8h..
      bfr[j] = frag_join(lo, hi);
    }
#pragma unroll
    for (int i = 0; i < 2; ++i)
#pragma unroll
      for (int j = 0; j < 2; ++j)
        acc[i][j] = __builtin_amdgcn_wmma_f32_16x16x32_f16(
            false, af[i], false, bfr[j], (short)0, acc[i][j], false, false);
    __syncthreads();   // buffer may be overwritten 2 tiles later
  }

  // C/D layout: lane = N, vgpr element t -> M = t + 8*(lane>>4)
#pragma unroll
  for (int i = 0; i < 2; ++i) {
#pragma unroll
    for (int j = 0; j < 2; ++j) {
      const int col   = blockN + wn + j * 16 + lr;
      const int rbase = blockM + wm + i * 16 + 8 * lh;
#pragma unroll
      for (int t = 0; t < 8; ++t)
        C[(size_t)(rbase + t) * ldc + col] = acc[i][j][t] * alpha;
    }
  }
}

// ---------------------------------------------------------------------------
// One-shot fp32 -> f16 conversion (straight and transposing) feeders.
// ---------------------------------------------------------------------------
__global__ void __launch_bounds__(256) cvt_f16_kernel(
    const float* __restrict__ src, _Float16* __restrict__ dst, int n)
{
  const int i = blockIdx.x * 256 + threadIdx.x;
  if (i < n) dst[i] = (_Float16)src[i];
}

// src [R][Ccols] -> dst [Ccols][R]
__global__ void __launch_bounds__(256) cvt_tr_f16_kernel(
    const float* __restrict__ src, _Float16* __restrict__ dst, int R, int Ccols)
{
  const int i = blockIdx.x * 256 + threadIdx.x;
  if (i < R * Ccols) {
    const int r = i / Ccols, c = i % Ccols;
    dst[(size_t)c * R + r] = (_Float16)src[i];
  }
}

// ---------------------------------------------------------------------------
// q_mean / k_mean over heads (q = qkv cols [0,512), k = cols [512,1024)).
// ---------------------------------------------------------------------------
__global__ void __launch_bounds__(256) head_mean_kernel(
    const float* __restrict__ qkv, float* __restrict__ qmean,
    float* __restrict__ kmean)
{
  const int i = blockIdx.x * 256 + threadIdx.x;   // 0 .. S_LEN*HD
  const int s = i >> 6, j = i & 63;
  const float* row = qkv + (size_t)s * (3 * D_MODEL);
  float a = 0.f, b = 0.f;
#pragma unroll
  for (int h = 0; h < NH; ++h) {
    a += row[h * HD + j];
    b += row[D_MODEL + h * HD + j];
  }
  qmean[i] = a * (1.0f / NH);
  kmean[i] = b * (1.0f / NH);
}

// ---------------------------------------------------------------------------
// Top-K=128 of a 2048-wide score row: order-preserving uint remap + 32-step
// bitwise threshold search (wave shuffle reduce + LDS atomic), then emit
// the >thr set and fill with ==thr. Downstream softmax+sum is permutation
// invariant, so only the selected set matters.
// ---------------------------------------------------------------------------
__global__ void __launch_bounds__(256) topk_kernel(
    const float* __restrict__ scores, int* __restrict__ sel)
{
  __shared__ int cnt;
  __shared__ int pos;
  const int s   = blockIdx.x;
  const int tid = threadIdx.x;
  const float* row = scores + (size_t)s * S_LEN;

  unsigned u[8];
#pragma unroll
  for (int i = 0; i < 8; ++i) {
    const unsigned b = __float_as_uint(row[tid + 256 * i]);
    u[i] = (b & 0x80000000u) ? ~b : (b | 0x80000000u);
  }

  unsigned thr = 0u;
  for (int bit = 31; bit >= 0; --bit) {
    const unsigned cand = thr | (1u << bit);
    if (tid == 0) cnt = 0;
    __syncthreads();
    int local = 0;
#pragma unroll
    for (int i = 0; i < 8; ++i) local += (u[i] >= cand) ? 1 : 0;
#pragma unroll
    for (int o = 16; o > 0; o >>= 1) local += __shfl_xor(local, o, 32);
    if ((tid & 31) == 0) atomicAdd(&cnt, local);
    __syncthreads();
    if (cnt >= TOPK) thr = cand;
    __syncthreads();
  }

  if (tid == 0) pos = 0;
  __syncthreads();
#pragma unroll
  for (int i = 0; i < 8; ++i) {
    if (u[i] > thr) {
      const int p = atomicAdd(&pos, 1);
      if (p < TOPK) sel[(size_t)s * TOPK + p] = tid + 256 * i;
    }
  }
  __syncthreads();
#pragma unroll
  for (int i = 0; i < 8; ++i) {
    if (u[i] == thr) {
      const int p = atomicAdd(&pos, 1);
      if (p < TOPK) sel[(size_t)s * TOPK + p] = tid + 256 * i;
    }
  }
}

// ---------------------------------------------------------------------------
// Gathered sparse attention. One block per query s, one wave per head.
// ---------------------------------------------------------------------------
__global__ void __launch_bounds__(256) sparse_attn_kernel(
    const float* __restrict__ qkv, const int* __restrict__ sel,
    float* __restrict__ attnb)
{
  __shared__ int   sidx[TOPK];
  __shared__ float sq[NH][HD];
  __shared__ float sattn[NH][TOPK];

  const int s    = blockIdx.x;
  const int tid  = threadIdx.x;
  const int lane = tid & 31;
  const int h    = tid >> 5;

  if (tid < TOPK) sidx[tid] = sel[(size_t)s * TOPK + tid];
  {
    const float2 qv =
        *(const float2*)(qkv + (size_t)s * (3 * D_MODEL) + h * HD + lane * 2);
    sq[h][lane * 2]     = qv.x;
    sq[h][lane * 2 + 1] = qv.y;
  }
  __syncthreads();

  float sc[4];
#pragma unroll
  for (int t = 0; t < 4; ++t) {
    const int kk  = lane * 4 + t;
    const int idx = sidx[kk];
    const float* kp = qkv + (size_t)idx * (3 * D_MODEL) + D_MODEL + h * HD;
    float d = 0.f;
#pragma unroll
    for (int j = 0; j < HD; j += 4) {
      const float4 kv = *(const float4*)(kp + j);
      const float4 qv = *(const float4*)(&sq[h][j]);
      d += kv.x * qv.x + kv.y * qv.y + kv.z * qv.z + kv.w * qv.w;
    }
    sc[t] = d * 0.125f;   // 1/sqrt(64)
  }

  float mx = fmaxf(fmaxf(sc[0], sc[1]), fmaxf(sc[2], sc[3]));
#pragma unroll
  for (int o = 16; o > 0; o >>= 1) mx = fmaxf(mx, __shfl_xor(mx, o, 32));
  float e[4], sum = 0.f;
#pragma unroll
  for (int t = 0; t < 4; ++t) { e[t] = __expf(sc[t] - mx); sum += e[t]; }
#pragma unroll
  for (int o = 16; o > 0; o >>= 1) sum += __shfl_xor(sum, o, 32);
  const float inv = 1.0f / sum;
#pragma unroll
  for (int t = 0; t < 4; ++t) sattn[h][lane * 4 + t] = e[t] * inv;
  __syncthreads();

  float a0 = 0.f, a1 = 0.f;
  for (int kk = 0; kk < TOPK; ++kk) {
    const int idx = sidx[kk];
    const float w = sattn[h][kk];
    const float2 vv = *(const float2*)(qkv + (size_t)idx * (3 * D_MODEL) +
                                       2 * D_MODEL + h * HD + lane * 2);
    a0 += w * vv.x;
    a1 += w * vv.y;
  }
  attnb[(size_t)s * D_MODEL + h * HD + lane * 2]     = a0;
  attnb[(size_t)s * D_MODEL + h * HD + lane * 2 + 1] = a1;
}

// ---------------------------------------------------------------------------
// Launcher. Inputs (setup_inputs order): x, Wqkv, Wo, Wqi, Wki — fp32.
// Output: [2048, 512] fp32.
// Workspace: fp32 { qkv, qmean, kmean, qi, ki, iscores, attnb } + sel(int)
//          + f16  { xh, Wqkvt, Wot, Wqit, Wkit, qmh, kmh, qih, kih, attnh }
// ~44 MB total.
// ---------------------------------------------------------------------------
extern "C" void kernel_launch(void* const* d_in, const int* in_sizes, int n_in,
                              void* d_out, int out_size, void* d_ws,
                              size_t ws_size, hipStream_t stream) {
  (void)in_sizes; (void)n_in; (void)out_size; (void)ws_size;
  const float* x    = (const float*)d_in[0];
  const float* Wqkv = (const float*)d_in[1];
  const float* Wo   = (const float*)d_in[2];
  const float* Wqi  = (const float*)d_in[3];
  const float* Wki  = (const float*)d_in[4];
  float* out = (float*)d_out;

  float* ws    = (float*)d_ws;
  float* qkv   = ws;
  float* qmean = qkv   + (size_t)S_LEN * 3 * D_MODEL;
  float* kmean = qmean + (size_t)S_LEN * HD;
  float* qi    = kmean + (size_t)S_LEN * HD;
  float* ki    = qi    + (size_t)S_LEN * RANK;
  float* isc   = ki    + (size_t)S_LEN * RANK;
  float* attnb = isc   + (size_t)S_LEN * S_LEN;
  int*   sel   = (int*)(attnb + (size_t)S_LEN * D_MODEL);

  _Float16* xh     = (_Float16*)(sel + (size_t)S_LEN * TOPK);
  _Float16* Wqkvt  = xh    + (size_t)S_LEN * D_MODEL;          // [1536][512]
  _Float16* Wot    = Wqkvt + (size_t)3 * D_MODEL * D_MODEL;    // [512][512]
  _Float16* Wqit   = Wot   + (size_t)D_MODEL * D_MODEL;        // [64][64]
  _Float16* Wkit   = Wqit  + (size_t)HD * RANK;
  _Float16* qmh    = Wkit  + (size_t)HD * RANK;
  _Float16* kmh    = qmh   + (size_t)S_LEN * HD;
  _Float16* qih    = kmh   + (size_t)S_LEN * HD;
  _Float16* kih    = qih   + (size_t)S_LEN * RANK;
  _Float16* attnh  = kih   + (size_t)S_LEN * RANK;

  const int G = 256;
  // 0) one-shot f16 conversion / weight transposition
  cvt_f16_kernel<<<(S_LEN * D_MODEL + G - 1) / G, G, 0, stream>>>(
      x, xh, S_LEN * D_MODEL);
  cvt_tr_f16_kernel<<<(3 * D_MODEL * D_MODEL + G - 1) / G, G, 0, stream>>>(
      Wqkv, Wqkvt, D_MODEL, 3 * D_MODEL);
  cvt_tr_f16_kernel<<<(D_MODEL * D_MODEL + G - 1) / G, G, 0, stream>>>(
      Wo, Wot, D_MODEL, D_MODEL);
  cvt_tr_f16_kernel<<<(HD * RANK + G - 1) / G, G, 0, stream>>>(
      Wqi, Wqit, HD, RANK);
  cvt_tr_f16_kernel<<<(HD * RANK + G - 1) / G, G, 0, stream>>>(
      Wki, Wkit, HD, RANK);

  // 1) qkv = x @ Wqkv
  wmma_gemm_f16<<<dim3(3 * D_MODEL / 64, S_LEN / 128), 256, 0, stream>>>(
      xh, Wqkvt, qkv, S_LEN, 3 * D_MODEL, D_MODEL,
      D_MODEL, D_MODEL, 3 * D_MODEL, 1.0f);

  // 2) head means + f16 feeders
  head_mean_kernel<<<(S_LEN * HD) / G, G, 0, stream>>>(qkv, qmean, kmean);
  cvt_f16_kernel<<<(S_LEN * HD + G - 1) / G, G, 0, stream>>>(qmean, qmh, S_LEN * HD);
  cvt_f16_kernel<<<(S_LEN * HD + G - 1) / G, G, 0, stream>>>(kmean, kmh, S_LEN * HD);

  // 3) qi = q_mean @ Wqi ; ki = k_mean @ Wki
  wmma_gemm_f16<<<dim3(RANK / 64, S_LEN / 128), 256, 0, stream>>>(
      qmh, Wqit, qi, S_LEN, RANK, HD, HD, HD, RANK, 1.0f);
  wmma_gemm_f16<<<dim3(RANK / 64, S_LEN / 128), 256, 0, stream>>>(
      kmh, Wkit, ki, S_LEN, RANK, HD, HD, HD, RANK, 1.0f);
  cvt_f16_kernel<<<(S_LEN * RANK + G - 1) / G, G, 0, stream>>>(qi, qih, S_LEN * RANK);
  cvt_f16_kernel<<<(S_LEN * RANK + G - 1) / G, G, 0, stream>>>(ki, kih, S_LEN * RANK);

  // 4) idx_scores = qi @ ki^T * 1/sqrt(R)   (kih is already [t][r] = B^T form)
  wmma_gemm_f16<<<dim3(S_LEN / 64, S_LEN / 128), 256, 0, stream>>>(
      qih, kih, isc, S_LEN, S_LEN, RANK, RANK, RANK, S_LEN, 0.125f);

  // 5) top-k indices per query
  topk_kernel<<<S_LEN, 256, 0, stream>>>(isc, sel);

  // 6) gathered sparse attention
  sparse_attn_kernel<<<S_LEN, 256, 0, stream>>>(qkv, sel, attnb);

  // 7) out = attnb @ Wo
  cvt_f16_kernel<<<(S_LEN * D_MODEL + G - 1) / G, G, 0, stream>>>(
      attnb, attnh, S_LEN * D_MODEL);
  wmma_gemm_f16<<<dim3(D_MODEL / 64, S_LEN / 128), 256, 0, stream>>>(
      attnh, Wot, out, S_LEN, D_MODEL, D_MODEL,
      D_MODEL, D_MODEL, D_MODEL, 1.0f);
}
